// LRNO_66254165508922
// MI455X (gfx1250) — matrix operand order repaired and checked by reading, not compiled, not run on hardware
//
#include <hip/hip_runtime.h>
#include <hip/hip_bf16.h>
#include <math.h>

// ---------------------------------------------------------------------------
// LRNO (low-rank neural operator) for MI455X / gfx1250, wave32 + WMMA.
//
// Algebraic restructuring (exact, from einsum factorization):
//   temp[b,n,o]   = sum_i h[b,n,i] * (Z3psi[b,n,:] @ W4r[:, o*32+i] + b4r)
//   tsum[b,o]     = sum_n temp[b,n,o]
//   out1[b,m,o]   = tsum[b,o] * (Z3phi[b,m,:] @ W4p[:,o] + b4p[o]) / 1024
//   h_new         = out1 + h @ W_i + b_i
// where W4r[k,c]=sum_r W4[k,c*4+r] (256x1024) and W4p[k,o]=sum_j W4[k,o*128+j].
//
// B-operand weights pre-shuffled into WMMA-fragment layout: 32x16 tile stored
// as [tileIdx][lane][16 contiguous f16], tileIdx = nt*(K/32)+kk, so a B
// fragment is two coalesced 16B loads per lane.  Hot GEMM loops hold one
// operand set resident in VGPRs (A frags in stages 2/3; B frags in stage 4)
// so each B (weight) tile is fetched exactly once per wave.
// ---------------------------------------------------------------------------

typedef __attribute__((ext_vector_type(16))) _Float16 v16h;
typedef __attribute__((ext_vector_type(8)))  _Float16 v8h;
typedef __attribute__((ext_vector_type(8)))  float    v8f;

// Branch-free gelu (tanh form): x * sigmoid(1.5957691*(x + 0.044715 x^3)).
// Lowers to FMAs + one v_exp_f32 (TRANS co-executes with WMMA), no branches.
__device__ __forceinline__ float gelu_f(float x) {
  const float u = 1.5957691216057308f * (x + 0.044715f * x * x * x);
  return __fdividef(x, 1.0f + __expf(-u));
}

__device__ __forceinline__ v8f wmma_f16(v16h a, v16h b, v8f c) {
  // D = A(16x32 f16) * B(32x16 f16) + C(16x16 f32)
  return __builtin_amdgcn_wmma_f32_16x16x32_f16(
      false, a, false, b, (short)0, c, false, false);
}

__device__ __forceinline__ v16h make_v16h(v8h lo, v8h hi) {
  return __builtin_shufflevector(lo, hi, 0, 1, 2, 3, 4, 5, 6, 7,
                                 8, 9, 10, 11, 12, 13, 14, 15);
}

// A fragment 16x32 (f16, row-major src in LDS, stride ld halves, 16B-aligned).
// lanes 0-15: rows 0-15, K {0-7,16-23}; lanes 16-31: K {8-15,24-31}.
// Two contiguous 16B vector loads per lane -> ds_load_b128.
__device__ __forceinline__ v16h load_a(const _Float16* src, int ld) {
  const int lane = threadIdx.x & 31;
  const _Float16* p = src + (lane & 15) * ld + (lane >> 4) * 8;
  const v8h lo = *(const v8h*)(p);
  const v8h hi = *(const v8h*)(p + 16);
  return make_v16h(lo, hi);
}

// B fragment from pre-shuffled weights: 32B contiguous per lane.
__device__ __forceinline__ v16h load_bs(const _Float16* src, int tileIdx) {
  const int lane = threadIdx.x & 31;
  const _Float16* p = src + tileIdx * 512 + lane * 16;
  const v8h lo = *(const v8h*)(p);
  const v8h hi = *(const v8h*)(p + 8);
  return make_v16h(lo, hi);
}

// ---------------------------------------------------------------------------
// Shared front: Z1=gelu(a@W1+b1) 64x64, Z2=gelu(Z1@W2+b2) 64x128,
// Z3=gelu(Z2@W3+b3) 64x256.  Z1,Z3 live in sA (32KB), Z2 in sB (16KB).
// Block = 256 threads = 8 waves, covers 64 consecutive points.
// W2s: K=64 (KT=2), N=128; W3s: K=128 (KT=4), N=256  (shuffled layouts).
// ---------------------------------------------------------------------------
__device__ __forceinline__ void dense_front(
    _Float16* sA, _Float16* sB, const float* Apts,
    const float* W1, const float* b1,
    const _Float16* W2s, const float* b2,
    const _Float16* W3s, const float* b3, int p0) {
  const int t = threadIdx.x;
  const int lane = t & 31;
  const int w = t >> 5;

  // ---- stage 1 (VALU, K=2): Z1 -> sA[0 .. 64*64)
  {
    const int c  = t & 63;
    const int pb = (t >> 6) * 16;
    const float w0 = W1[c], w1 = W1[64 + c], bc = b1[c];
#pragma unroll
    for (int q = 0; q < 16; ++q) {
      const int pl = pb + q;
      const float a0 = Apts[(p0 + pl) * 2 + 0];
      const float a1 = Apts[(p0 + pl) * 2 + 1];
      sA[pl * 64 + c] = (_Float16)gelu_f(a0 * w0 + a1 * w1 + bc);
    }
  }
  __syncthreads();

  // ---- stage 2 (WMMA): Z2 (64x128) -> sB.  mt fixed per wave; A resident.
  {
    const int mt = w >> 1;
    const v16h a0 = load_a(sA + mt * 16 * 64 + 0, 64);
    const v16h a1 = load_a(sA + mt * 16 * 64 + 32, 64);
    for (int jt = 0; jt < 4; ++jt) {
      const int nt = (w & 1) * 4 + jt;
      v8f acc = {};
      acc = wmma_f16(a0, load_bs(W2s, nt * 2 + 0), acc);
      acc = wmma_f16(a1, load_bs(W2s, nt * 2 + 1), acc);
      const int col = nt * 16 + (lane & 15);
      const float bias = b2[col];
#pragma unroll
      for (int j = 0; j < 8; ++j) {
        const int row = mt * 16 + j + 8 * (lane >> 4);
        sB[row * 128 + col] = (_Float16)gelu_f(acc[j] + bias);
      }
    }
  }
  __syncthreads();

  // ---- stage 3 (WMMA): Z3 (64x256) -> sA.  mt fixed per wave; A resident.
  {
    const int mt = w >> 1;
    v16h af[4];
#pragma unroll
    for (int kk = 0; kk < 4; ++kk)
      af[kk] = load_a(sB + mt * 16 * 128 + kk * 32, 128);
    for (int jt = 0; jt < 8; ++jt) {
      const int nt = (w & 1) * 8 + jt;
      v8f acc = {};
#pragma unroll
      for (int kk = 0; kk < 4; ++kk)
        acc = wmma_f16(af[kk], load_bs(W3s, nt * 4 + kk), acc);
      const int col = nt * 16 + (lane & 15);
      const float bias = b3[col];
#pragma unroll
      for (int j = 0; j < 8; ++j) {
        const int row = mt * 16 + j + 8 * (lane >> 4);
        sA[row * 256 + col] = (_Float16)gelu_f(acc[j] + bias);
      }
    }
  }
  __syncthreads();
}

// ---------------------------------------------------------------------------
// psi kernel: dense front + fused psiR GEMM (K=256 -> N=1024), x h[p,i],
// reduce over n, i (regs + shfl) and i halves (LDS atomics) -> global tsum[b,o] atomics.
// nt outer with B fragments resident in VGPRs (each B tile loaded once);
// partial sums accumulated across the 4 M-tiles before the lane reduce.
// ---------------------------------------------------------------------------
__global__ void __launch_bounds__(256) k_psi(
    const float* __restrict__ Apts,
    const float* __restrict__ W1, const float* __restrict__ b1,
    const _Float16* __restrict__ W2s, const float* __restrict__ b2,
    const _Float16* __restrict__ W3s, const float* __restrict__ b3,
    const _Float16* __restrict__ W4s, const float* __restrict__ b4r,
    const float* __restrict__ hcur, float* __restrict__ tsum) {
  __shared__ __align__(16) _Float16 sA[64 * 256];
  __shared__ __align__(16) _Float16 sB[64 * 128];
  __shared__ float sH[64 * 32];
  __shared__ float sTsum[32];

  const int t = threadIdx.x;
  const int lane = t & 31;
  const int w = t >> 5;
  const int p0 = blockIdx.x * 64;
  const int bb = blockIdx.x >> 4;  // batch (16 blocks per batch)

#pragma unroll
  for (int q = 0; q < 8; ++q) sH[t + q * 256] = hcur[p0 * 32 + t + q * 256];
  if (t < 32) sTsum[t] = 0.0f;

  dense_front(sA, sB, Apts, W1, b1, W2s, b2, W3s, b3, p0);

  // stage 4: 256 tiles (4 M x 64 N).  Wave w owns nt = w*8..w*8+7.
  for (int jn = 0; jn < 8; ++jn) {
    const int nt = w * 8 + jn;
    v16h bf[8];
#pragma unroll
    for (int kk = 0; kk < 8; ++kk) bf[kk] = load_bs(W4s, nt * 8 + kk);

    const int col = nt * 16 + (lane & 15);
    const int ii  = col & 31;  // h-channel index
    const float bias = b4r[col];
    float ssum = 0.0f;
#pragma unroll
    for (int mt = 0; mt < 4; ++mt) {
      v8f acc = {};
#pragma unroll
      for (int kk = 0; kk < 8; ++kk) {
        v16h a = load_a(sA + mt * 16 * 256 + kk * 32, 256);
        acc = wmma_f16(a, bf[kk], acc);
      }
#pragma unroll
      for (int j = 0; j < 8; ++j) {
        const int row = mt * 16 + j + 8 * (lane >> 4);
        ssum += (acc[j] + bias) * sH[row * 32 + ii];
      }
    }
    ssum += __shfl_xor(ssum, 16, 32);
    ssum += __shfl_xor(ssum, 8, 32);
    ssum += __shfl_xor(ssum, 4, 32);
    ssum += __shfl_xor(ssum, 2, 32);
    ssum += __shfl_xor(ssum, 1, 32);
    if (lane == 0) atomicAdd(&sTsum[nt >> 1], ssum);
  }
  __syncthreads();
  if (t < 32) atomicAdd(&tsum[bb * 32 + t], sTsum[t]);
}

// ---------------------------------------------------------------------------
// phi kernel: dense front + psum GEMM (K=256 -> N=32), epilogue fuses
// h_new = tsum*psum/1024 + h@Wi + bi.
// ---------------------------------------------------------------------------
__global__ void __launch_bounds__(256) k_phi(
    const float* __restrict__ Apts,
    const float* __restrict__ W1, const float* __restrict__ b1,
    const _Float16* __restrict__ W2s, const float* __restrict__ b2,
    const _Float16* __restrict__ W3s, const float* __restrict__ b3,
    const _Float16* __restrict__ W4s, const float* __restrict__ b4p,
    const float* __restrict__ tsum, const float* __restrict__ hcur,
    const float* __restrict__ wW, const float* __restrict__ wB,
    float* __restrict__ hnext) {
  __shared__ __align__(16) _Float16 sA[64 * 256];
  __shared__ __align__(16) _Float16 sB[64 * 128];
  __shared__ float sH[64 * 32];

  const int t = threadIdx.x;
  const int lane = t & 31;
  const int w = t >> 5;
  const int p0 = blockIdx.x * 64;
  const int bb = blockIdx.x >> 4;

#pragma unroll
  for (int q = 0; q < 8; ++q) sH[t + q * 256] = hcur[p0 * 32 + t + q * 256];

  dense_front(sA, sB, Apts, W1, b1, W2s, b2, W3s, b3, p0);

  // stage 4: 8 tiles (4 M x 2 N), one per wave.  K=256 (KT=8), N=32.
  {
    const int mt = w >> 1, nt = w & 1;
    v8f acc = {};
#pragma unroll
    for (int kk = 0; kk < 8; ++kk) {
      v16h a  = load_a(sA + mt * 16 * 256 + kk * 32, 256);
      v16h bf = load_bs(W4s, nt * 8 + kk);
      acc = wmma_f16(a, bf, acc);
    }
    const int col = nt * 16 + (lane & 15);
    const float bias = b4p[col];
    const float ts = tsum[bb * 32 + col] * (1.0f / 1024.0f);
    const float bw = wB[col];
#pragma unroll
    for (int j = 0; j < 8; ++j) {
      const int row = mt * 16 + j + 8 * (lane >> 4);
      float hw = bw;
#pragma unroll
      for (int i = 0; i < 32; ++i) hw += sH[row * 32 + i] * wW[i * 32 + col];
      hnext[(p0 + row) * 32 + col] = ts * (acc[j] + bias) + hw;
    }
  }
}

// ---------------------------------------------------------------------------
// Prep / conversion kernels
// ---------------------------------------------------------------------------
__global__ void k_points(float* __restrict__ A, float* __restrict__ h0,
                         float* __restrict__ tsum, const float* __restrict__ v,
                         const float* __restrict__ fc0W,
                         const float* __restrict__ fc0b) {
  const int p = blockIdx.x * 256 + threadIdx.x;
  if (p < 1024) tsum[p] = 0.0f;
  if (p >= 8192) return;
  const int b = p >> 10, s = p & 1023;
  const float a0 = v[b * 1032 + 8 + s];
  const float a1 = (float)s * (1.0f / 1023.0f);
  A[p * 2] = a0;
  A[p * 2 + 1] = a1;
#pragma unroll
  for (int o = 0; o < 32; ++o)
    h0[p * 32 + o] = a0 * fc0W[o] + a1 * fc0W[32 + o] + fc0b[o];
}

// Cast f32 (K x N row-major) -> f16 in WMMA-fragment-shuffled layout.
__global__ void k_shuffle_cast(_Float16* __restrict__ dst,
                               const float* __restrict__ src, int K, int N) {
  const int id = blockIdx.x * 256 + threadIdx.x;
  if (id >= K * N) return;
  const int hh = id & 15;
  const int lane = (id >> 4) & 31;
  const int tileIdx = id >> 9;
  const int KT = K >> 5;
  const int kk = tileIdx % KT;
  const int nt = tileIdx / KT;
  const int k = kk * 32 + (lane >> 4) * 8 + (hh & 7) + (hh >> 3) * 16;
  const int n = nt * 16 + (lane & 15);
  dst[id] = (_Float16)src[k * N + n];
}

// psi fold (sum over r) + shuffle: effective 256 x 1024 f16, KT=8.
__global__ void k_reduce_psi(_Float16* __restrict__ dstW,
                             float* __restrict__ dstB,
                             const float* __restrict__ W4,
                             const float* __restrict__ b4) {
  const int id = blockIdx.x * 256 + threadIdx.x;
  if (id < 256 * 1024) {
    const int hh = id & 15;
    const int lane = (id >> 4) & 31;
    const int tileIdx = id >> 9;
    const int kk = tileIdx & 7;      // KT=8
    const int nt = tileIdx >> 3;
    const int k = kk * 32 + (lane >> 4) * 8 + (hh & 7) + (hh >> 3) * 16;
    const int c = nt * 16 + (lane & 15);
    const float* s = W4 + k * 4096 + c * 4;
    dstW[id] = (_Float16)(s[0] + s[1] + s[2] + s[3]);
  }
  if (id < 1024) {
    const float* s = b4 + id * 4;
    dstB[id] = s[0] + s[1] + s[2] + s[3];
  }
}

// phi fold (sum over 128) + shuffle: effective 256 x 32 f16, KT=8.
__global__ void k_reduce_phi(_Float16* __restrict__ dstW,
                             float* __restrict__ dstB,
                             const float* __restrict__ W4,
                             const float* __restrict__ b4) {
  const int id = blockIdx.x * 256 + threadIdx.x;
  if (id < 256 * 32) {
    const int hh = id & 15;
    const int lane = (id >> 4) & 31;
    const int tileIdx = id >> 9;
    const int kk = tileIdx & 7;      // KT=8
    const int nt = tileIdx >> 3;
    const int k = kk * 32 + (lane >> 4) * 8 + (hh & 7) + (hh >> 3) * 16;
    const int o = nt * 16 + (lane & 15);
    const float* s = W4 + k * 4096 + o * 128;
    float acc = 0.0f;
    for (int j = 0; j < 128; ++j) acc += s[j];
    dstW[id] = (_Float16)acc;
  }
  if (id < 32) {
    const float* s = b4 + id * 128;
    float acc = 0.0f;
    for (int j = 0; j < 128; ++j) acc += s[j];
    dstB[id] = acc;
  }
}

// final head: out = (h @ fc1 + b1) @ fc2 + b2   (no gelu)
__global__ void k_final(const float* __restrict__ hcur,
                        const float* __restrict__ fc1W,
                        const float* __restrict__ fc1b,
                        const float* __restrict__ fc2W,
                        const float* __restrict__ fc2b,
                        float* __restrict__ out) {
  const int p = blockIdx.x * 256 + threadIdx.x;
  if (p >= 8192) return;
  float h[32];
#pragma unroll
  for (int i = 0; i < 32; ++i) h[i] = hcur[p * 32 + i];
  float o = fc2b[0];
  for (int j = 0; j < 64; ++j) {
    float t1 = fc1b[j];
#pragma unroll
    for (int i = 0; i < 32; ++i) t1 += h[i] * fc1W[i * 64 + j];
    o += t1 * fc2W[j];
  }
  out[p] = o;
}

// ---------------------------------------------------------------------------
// Host orchestration.
// Input order assumption (setup_inputs dict order; params pytree sorted keys):
//   0: v
//   1: fc0.W  2: fc0.b  3: fc1.W  4: fc1.b  5: fc2.W  6: fc2.b
//   7+8i+l  : phi[i] layer leaves (W1,b1,W2,b2,W3,b3,W4,b4)
//   39+8i+l : psi[i] layer leaves
//   71+2i / 72+2i : w[i].W / w[i].b
// ---------------------------------------------------------------------------
extern "C" void kernel_launch(void* const* d_in, const int* in_sizes, int n_in,
                              void* d_out, int out_size, void* d_ws,
                              size_t ws_size, hipStream_t stream) {
  (void)in_sizes; (void)n_in; (void)out_size; (void)ws_size;
  const float* v    = (const float*)d_in[0];
  const float* fc0W = (const float*)d_in[1];
  const float* fc0b = (const float*)d_in[2];
  const float* fc1W = (const float*)d_in[3];
  const float* fc1b = (const float*)d_in[4];
  const float* fc2W = (const float*)d_in[5];
  const float* fc2b = (const float*)d_in[6];
  auto PHI = [&](int i, int l) { return (const float*)d_in[7 + 8 * i + l]; };
  auto PSI = [&](int i, int l) { return (const float*)d_in[39 + 8 * i + l]; };
  auto WW  = [&](int i) { return (const float*)d_in[71 + 2 * i]; };
  auto WB  = [&](int i) { return (const float*)d_in[72 + 2 * i]; };

  char* ws = (char*)d_ws;
  float*     A    = (float*)(ws + 0);                 // 8192*2 f32
  float*     hA   = (float*)(ws + 65536);             // 8192*32 f32
  float*     hB   = (float*)(ws + 1114112);           // 8192*32 f32
  float*     tsum = (float*)(ws + 2162688);           // 4*8*32 f32
  _Float16*  W2s  = (_Float16*)(ws + 2166784);        // 8 nets * 64*128
  _Float16*  W3s  = (_Float16*)(ws + 2297856);        // 8 nets * 128*256
  _Float16*  W4s  = (_Float16*)(ws + 2822144);        // 4 nets * 256*1024
  float*     B4r  = (float*)(ws + 4919296);           // 4 * 1024 f32
  _Float16*  W4p  = (_Float16*)(ws + 4935680);        // 4 nets * 256*32
  float*     B4p  = (float*)(ws + 5001216);           // 4 * 32 f32

  // prep: points, h0, tsum zero
  k_points<<<32, 256, 0, stream>>>(A, hA, tsum, v, fc0W, fc0b);

  // weight conversion (fragment-shuffled): nets 0-3 = psi, 4-7 = phi
  for (int i = 0; i < 4; ++i) {
    k_shuffle_cast<<<32, 256, 0, stream>>>(W2s + i * 8192, PSI(i, 2), 64, 128);
    k_shuffle_cast<<<128, 256, 0, stream>>>(W3s + i * 32768, PSI(i, 4), 128, 256);
    k_shuffle_cast<<<32, 256, 0, stream>>>(W2s + (4 + i) * 8192, PHI(i, 2), 64, 128);
    k_shuffle_cast<<<128, 256, 0, stream>>>(W3s + (4 + i) * 32768, PHI(i, 4), 128, 256);
    k_reduce_psi<<<1024, 256, 0, stream>>>(W4s + i * 262144, B4r + i * 1024,
                                           PSI(i, 6), PSI(i, 7));
    k_reduce_phi<<<32, 256, 0, stream>>>(W4p + i * 8192, B4p + i * 32,
                                         PHI(i, 6), PHI(i, 7));
  }

  float* hcur = hA;
  float* hnext = hB;
  for (int it = 0; it < 4; ++it) {
    k_psi<<<128, 256, 0, stream>>>(
        A, PSI(it, 0), PSI(it, 1), W2s + it * 8192, PSI(it, 3),
        W3s + it * 32768, PSI(it, 5), W4s + it * 262144, B4r + it * 1024,
        hcur, tsum + it * 256);
    k_phi<<<128, 256, 0, stream>>>(
        A, PHI(it, 0), PHI(it, 1), W2s + (4 + it) * 8192, PHI(it, 3),
        W3s + (4 + it) * 32768, PHI(it, 5), W4p + it * 8192, B4p + it * 32,
        tsum + it * 256, hcur, WW(it), WB(it), hnext);
    float* tmp = hcur; hcur = hnext; hnext = tmp;
  }

  k_final<<<32, 256, 0, stream>>>(hcur, fc1W, fc1b, fc2W, fc2b, (float*)d_out);
}
